// Translator_26568667693227
// MI455X (gfx1250) — matrix operand order repaired and checked by reading, not compiled, not run on hardware
//
#include <hip/hip_runtime.h>
#include <hip/hip_bf16.h>
#include <math.h>

// ---------------- constants for this problem ----------------
#define BB_   8
#define CC_   256
#define HH_   64
#define WW_   64
#define SS_   256
#define LL_   3
#define MTOT  (BB_*HH_*WW_)      // 32768
#define NTOT  CC_                // 256
#define K3    (CC_*9)            // 2304
#define K1    CC_                // 256

#define LRELU_SLOPE 0.2f
#define S2_SCALE 1.1547005383792515f     // sqrt(4/3) = 2/sqrt(3)
#define INV_SQRT2 0.7071067811865476f
#define EPS_ 1e-8f

typedef __bf16 v16bf __attribute__((ext_vector_type(16)));
typedef __bf16 v8bf  __attribute__((ext_vector_type(8)));
typedef __bf16 v2bf  __attribute__((ext_vector_type(2)));
typedef float  v8f   __attribute__((ext_vector_type(8)));

// CDNA5 async global->LDS path (ASYNCcnt tracked), if this toolchain exposes it
#if __has_builtin(__builtin_amdgcn_global_load_async_to_lds_b128) && \
    __has_builtin(__builtin_amdgcn_s_wait_asynccnt)
#define USE_ASYNC_B 1
// per hipcc diagnostics: arg0 = v4i in AS(1), arg1 = v4i in AS(3), then imm offset, imm cpol
typedef int v4i_ __attribute__((vector_size(16)));
typedef __attribute__((address_space(1))) v4i_* as1_v4i_ptr;
typedef __attribute__((address_space(3))) v4i_* as3_v4i_ptr;
#else
#define USE_ASYNC_B 0
#endif

// ---------------- helpers ----------------
__device__ __forceinline__ float gauss_hash(unsigned seed, unsigned idx) {
    unsigned x = seed * 0x9E3779B9u ^ (idx + 0x7F4A7C15u);
    x ^= x >> 16; x *= 0x7FEB352Du; x ^= x >> 15; x *= 0x846CA68Bu; x ^= x >> 16;
    unsigned y = x * 0xC2B2AE35u ^ 0x68E31DA4u;
    y ^= y >> 16; y *= 0x2C1B3C6Du; y ^= y >> 15;
    float u1 = (float)(x >> 8) * (1.0f / 16777216.0f) + 1e-7f;
    float u2 = (float)(y >> 8) * (1.0f / 16777216.0f);
    return sqrtf(-2.0f * __logf(u1)) * __cosf(6.28318530718f * u2);
}

// ---------------- tiny prep kernels ----------------
__global__ void cvt_bf16_kernel(const float* __restrict__ src, __bf16* __restrict__ dst, int n) {
    int i = blockIdx.x * blockDim.x + threadIdx.x;
    if (i < n) dst[i] = (__bf16)src[i];
}

// mod[b,c] = sum_k s[b,k]*sW[c,k] + sb[c] + 1
__global__ void mod_kernel(const float* __restrict__ s, const float* __restrict__ sW,
                           const float* __restrict__ sb, float* __restrict__ mod) {
    int idx = blockIdx.x * blockDim.x + threadIdx.x;
    if (idx >= BB_ * CC_) return;
    int b = idx / CC_, c = idx % CC_;
    float acc = 0.f;
    for (int k = 0; k < SS_; ++k) acc += s[b * SS_ + k] * sW[c * SS_ + k];
    mod[idx] = acc + sb[c] + 1.0f;
}

// escale[b,o] = S2 * sqrt( sum_{i,t} (w[o,i,t]*mod[b,i])^2 + eps )
__global__ void demod_kernel(const float* __restrict__ w, const float* __restrict__ mod,
                             float* __restrict__ escale) {
    int bo = blockIdx.x;            // B*NTOT blocks
    int b = bo / NTOT, o = bo % NTOT;
    float sum = 0.f;
    for (int k = threadIdx.x; k < K3; k += blockDim.x) {
        int i = k / 9;
        float wm = w[(size_t)o * K3 + k] * mod[b * CC_ + i];
        sum += wm * wm;
    }
    __shared__ float red[256];
    red[threadIdx.x] = sum;
    __syncthreads();
    for (int st = 128; st > 0; st >>= 1) {
        if (threadIdx.x < st) red[threadIdx.x] += red[threadIdx.x + st];
        __syncthreads();
    }
    if (threadIdx.x == 0) escale[bo] = S2_SCALE * sqrtf(red[0] + EPS_);
}

// ---------------- WMMA implicit-GEMM conv kernel ----------------
// Y[m,n] = epilogue( sum_k A[m,k]*B[k,n] )
//  A[m,k] = mod[b,i] * X[b,i,oy+ty,ox+tx]   (im2col, KR x KR, pad KR/2)
//  B[k,n] = Wt[n,k]  (bf16, raw weights; [O][I*KR*KR] row-major)
// epilogue: v*escale[b,n] + bias[n] (+ nw[n]*noise(b,h,w)) (lrelu optional)
#define BM 128
#define BN 128
#define KC 32
#define LDA_STRIDE 40   // elements; 80B rows keep 16B alignment of 16B chunks

template <int KR, bool HASMOD, bool HASESC, bool HASNOISE, bool DOLRELU>
__global__ __launch_bounds__(256)
void gemm_conv_kernel(const float* __restrict__ X, const __bf16* __restrict__ Wt,
                      const float* __restrict__ mod, const float* __restrict__ escale,
                      const float* __restrict__ bias, const float* __restrict__ nw,
                      float* __restrict__ Y, int seed) {
    constexpr int K = (KR == 3) ? K3 : K1;
    __shared__ alignas(16) __bf16 lA[BM * LDA_STRIDE];
    __shared__ alignas(16) __bf16 lB[BN * LDA_STRIDE];

    const int tid   = threadIdx.x;
    const int lane  = tid & 31;
    const int wv    = tid >> 5;          // 8 waves
    const int wvM   = wv & 3;            // 4 along M (32 rows each)
    const int wvN   = wv >> 2;           // 2 along N (64 cols each)
    const int l16   = lane & 15;
    const int hi    = lane >> 4;         // K-half selector per ISA fragment layout

    const int mBlock = blockIdx.x * BM;
    const int nBlock = blockIdx.y * BN;

    v8f acc[2][4];
    #pragma unroll
    for (int tm = 0; tm < 2; ++tm)
        #pragma unroll
        for (int tn = 0; tn < 4; ++tn)
            acc[tm][tn] = (v8f){0.f,0.f,0.f,0.f,0.f,0.f,0.f,0.f};

    // ---- A staging: thread owns a pair of adjacent k values x 8 consecutive pixels
    const int kp  = tid & 15;            // k-pair index: k = k0 + 2*kp (+p)
    const int mc  = tid >> 4;            // m-chunk of 8 rows (16 chunks = 128 rows)
    const int m0s = mBlock + mc * 8;     // 8 consecutive pixels, same (b, oy)
    const int bS  = m0s >> 12;
    const int remS = m0s & 4095;
    const int oyS = remS >> 6;
    const int ox0 = remS & 63;           // multiple of 8

    // ---- B staging: thread owns one row-half (16 bf16) of the weight tile
    const int rB   = tid >> 1;
    const int halfB = tid & 1;

    const int KT = K / KC;
    for (int kt = 0; kt < KT; ++kt) {
        const int k0 = kt * KC;

        // ---- stage A tile (im2col gather; mod/div/tap hoisted per-k) ----
        {
            float va[2][8];
            #pragma unroll
            for (int p = 0; p < 2; ++p) {
                const int k = k0 + 2 * kp + p;
                float mv = 1.0f;
                if (KR == 3) {
                    const int i = k / 9;
                    const int tap = k - i * 9;
                    const int t3 = tap / 3;
                    const int dy = t3 - 1;
                    const int dx = tap - 3 * t3 - 1;
                    if (HASMOD) mv = mod[(bS << 8) + i];
                    const int iy = oyS + dy;
                    const bool rv = (unsigned)iy < 64u;
                    const float* Xp = X + (size_t)(((bS << 8) + i) << 12) + (iy << 6);
                    const int xb = ox0 + dx;
                    #pragma unroll
                    for (int q = 0; q < 8; ++q) {
                        const int ix = xb + q;
                        va[p][q] = (rv && (unsigned)ix < 64u) ? Xp[ix] * mv : 0.f;
                    }
                } else {
                    const int i = k;
                    if (HASMOD) mv = mod[(bS << 8) + i];
                    const float4* Xp4 =
                        (const float4*)(X + (size_t)(((bS << 8) + i) << 12) + (oyS << 6) + ox0);
                    float4 f0 = Xp4[0], f1 = Xp4[1];
                    va[p][0] = f0.x * mv; va[p][1] = f0.y * mv;
                    va[p][2] = f0.z * mv; va[p][3] = f0.w * mv;
                    va[p][4] = f1.x * mv; va[p][5] = f1.y * mv;
                    va[p][6] = f1.z * mv; va[p][7] = f1.w * mv;
                }
            }
            __bf16* dstBase = &lA[mc * 8 * LDA_STRIDE + 2 * kp];
            #pragma unroll
            for (int q = 0; q < 8; ++q) {      // packed pair -> single b32 LDS store
                v2bf pk;
                pk.x = (__bf16)va[0][q];
                pk.y = (__bf16)va[1][q];
                *(v2bf*)(dstBase + q * LDA_STRIDE) = pk;
            }
        }
        // ---- stage B tile: rows are output channels, K contiguous ----
        {
            const __bf16* src = Wt + (size_t)(nBlock + rB) * K + k0 + halfB * 16;
            __bf16* dst = &lB[rB * LDA_STRIDE + halfB * 16];
#if USE_ASYNC_B
            __builtin_amdgcn_global_load_async_to_lds_b128(
                (as1_v4i_ptr)(src), (as3_v4i_ptr)(dst), 0, 0);
            __builtin_amdgcn_global_load_async_to_lds_b128(
                (as1_v4i_ptr)(src + 8), (as3_v4i_ptr)(dst + 8), 0, 0);
#else
            uint4 u0 = *(const uint4*)(src);
            uint4 u1 = *(const uint4*)(src + 8);
            *(uint4*)dst       = u0;
            *(uint4*)(dst + 8) = u1;
#endif
            if (kt + 1 < KT) __builtin_prefetch(src + KC, 0, 1);
        }
#if USE_ASYNC_B
        __builtin_amdgcn_s_wait_asynccnt(0);
#endif
        __syncthreads();

        // ---- fragments per ISA layouts ----
        // A (16x32 bf16): lane<16 holds K {0..7,16..23}, lane>=16 holds K {8..15,24..31}, M = l16
        v16bf afr[2];
        #pragma unroll
        for (int tm = 0; tm < 2; ++tm) {
            int row = wvM * 32 + tm * 16 + l16;
            const __bf16* pa = &lA[row * LDA_STRIDE + (hi ? 8 : 0)];
            v8bf a0 = *(const v8bf*)pa;
            v8bf a1 = *(const v8bf*)(pa + 16);
            afr[tm] = __builtin_shufflevector(a0, a1, 0,1,2,3,4,5,6,7,8,9,10,11,12,13,14,15);
        }
        // B (32x16 bf16): lane<16 holds K {0..7,16..23}, lane>=16 holds K {8..15,24..31}, N = l16
        v16bf bfr[4];
        #pragma unroll
        for (int tn = 0; tn < 4; ++tn) {
            int rowN = wvN * 64 + tn * 16 + l16;
            const __bf16* pb = &lB[rowN * LDA_STRIDE + hi * 8];
            v8bf b0 = *(const v8bf*)pb;
            v8bf b1 = *(const v8bf*)(pb + 16);
            bfr[tn] = __builtin_shufflevector(b0, b1, 0,1,2,3,4,5,6,7,8,9,10,11,12,13,14,15);
        }
        #pragma unroll
        for (int tm = 0; tm < 2; ++tm)
            #pragma unroll
            for (int tn = 0; tn < 4; ++tn)
                acc[tm][tn] = __builtin_amdgcn_wmma_f32_16x16x32_bf16(
                    false, afr[tm], false, bfr[tn], (short)0, acc[tm][tn], false, false);
        __syncthreads();
    }

    // ---- epilogue: demod scale, bias, noise, lrelu; contiguous float4 stores ----
    #pragma unroll
    for (int tm = 0; tm < 2; ++tm) {
        int m0 = mBlock + wvM * 32 + tm * 16 + hi * 8;   // 8 consecutive pixels
        int b  = m0 >> 12;
        int r0 = m0 & 4095;
        int oy0 = r0 >> 6;
        int oxE = r0 & 63;
        #pragma unroll
        for (int tn = 0; tn < 4; ++tn) {
            int n = nBlock + wvN * 64 + tn * 16 + l16;
            float es  = HASESC ? escale[(b << 8) + n] : 1.0f;
            float bs  = bias[n];
            float nwv = HASNOISE ? nw[n] : 0.0f;
            v8f a = acc[tm][tn];
            float r[8];
            #pragma unroll
            for (int q = 0; q < 8; ++q) {
                float v = a[q] * es + bs;
                if (HASNOISE) v += nwv * gauss_hash((unsigned)seed, (unsigned)(m0 + q));
                if (DOLRELU) v = (v >= 0.f) ? v : LRELU_SLOPE * v;
                r[q] = v;
            }
            float* yp = Y + (size_t)(((b << 8) + n) << 12) + (oy0 << 6) + oxE;
            *(float4*)yp       = make_float4(r[0], r[1], r[2], r[3]);
            *((float4*)yp + 1) = make_float4(r[4], r[5], r[6], r[7]);
        }
    }
}

// ---------------- elementwise kernels ----------------
__global__ void residual_kernel(float* __restrict__ x, const float* __restrict__ h, int n) {
    int i = blockIdx.x * blockDim.x + threadIdx.x;
    if (i < n) x[i] = (x[i] + h[i]) * INV_SQRT2;
}

__global__ void final_kernel(const float* __restrict__ e, const float* __restrict__ mraw,
                             float* __restrict__ out_f, int n) {
    int i = blockIdx.x * blockDim.x + threadIdx.x;
    if (i < n) {
        float m = 1.0f / (1.0f + __expf(-mraw[i]));
        out_f[i] = out_f[i] * m + e[i] * (1.0f - m);
    }
}

// ---------------- host launch ----------------
extern "C" void kernel_launch(void* const* d_in, const int* in_sizes, int n_in,
                              void* d_out, int out_size, void* d_ws, size_t ws_size,
                              hipStream_t stream) {
    const float* e   = (const float*)d_in[0];
    const float* s   = (const float*)d_in[1];
    const float* w0  = (const float*)d_in[2];
    const float* b0  = (const float*)d_in[3];
    const float* bw  = (const float*)d_in[4];
    const float* bb  = (const float*)d_in[5];
    const float* bnw = (const float*)d_in[6];
    const float* bsW = (const float*)d_in[7];
    const float* bsb = (const float*)d_in[8];
    const float* wf  = (const float*)d_in[9];
    const float* bf  = (const float*)d_in[10];
    const float* sWf = (const float*)d_in[11];
    const float* sbf = (const float*)d_in[12];
    const float* wm  = (const float*)d_in[13];
    const float* bm  = (const float*)d_in[14];
    const float* sWm = (const float*)d_in[15];
    const float* sbm = (const float*)d_in[16];
    float* out = (float*)d_out;

    const size_t NEL = (size_t)BB_ * CC_ * HH_ * WW_;       // 8388608
    char* ws = (char*)d_ws;
    size_t off = 0;
    auto take = [&](size_t bytes) { char* p = ws + off; off += (bytes + 255) & ~(size_t)255; return p; };

    float*  xbuf  = (float*)take(NEL * 4);
    float*  hA    = (float*)take(NEL * 4);
    float*  hB    = (float*)take(NEL * 4);
    __bf16* wt0   = (__bf16*)take((size_t)CC_ * CC_ * 2);
    __bf16* wtB   = (__bf16*)take((size_t)6 * CC_ * K3 * 2);
    __bf16* wtF   = (__bf16*)take((size_t)CC_ * CC_ * 2);
    __bf16* wtM   = (__bf16*)take((size_t)CC_ * CC_ * 2);
    float*  modsB = (float*)take((size_t)6 * BB_ * CC_ * 4);
    float*  modF  = (float*)take((size_t)BB_ * CC_ * 4);
    float*  modM  = (float*)take((size_t)BB_ * CC_ * 4);
    float*  escB  = (float*)take((size_t)6 * BB_ * CC_ * 4);

    // 1) weight conversions to bf16 (B operand; already [O][K] row-major)
    {
        int n0 = CC_ * CC_;
        cvt_bf16_kernel<<<(n0 + 255) / 256, 256, 0, stream>>>(w0, wt0, n0);
        int nb = 6 * CC_ * K3;
        cvt_bf16_kernel<<<(nb + 255) / 256, 256, 0, stream>>>(bw, wtB, nb);
        cvt_bf16_kernel<<<(n0 + 255) / 256, 256, 0, stream>>>(wf, wtF, n0);
        cvt_bf16_kernel<<<(n0 + 255) / 256, 256, 0, stream>>>(wm, wtM, n0);
    }
    // 2) modulation vectors
    {
        int gm = (BB_ * CC_ + 255) / 256;
        for (int c = 0; c < 6; ++c)
            mod_kernel<<<gm, 256, 0, stream>>>(s, bsW + (size_t)c * CC_ * SS_,
                                               bsb + (size_t)c * CC_, modsB + (size_t)c * BB_ * CC_);
        mod_kernel<<<gm, 256, 0, stream>>>(s, sWf, sbf, modF);
        mod_kernel<<<gm, 256, 0, stream>>>(s, sWm, sbm, modM);
    }
    // 3) demod scales (faithful to source: multiply by sqrt(sum+eps)), S2 folded in
    for (int c = 0; c < 6; ++c)
        demod_kernel<<<BB_ * NTOT, 256, 0, stream>>>(bw + (size_t)c * CC_ * K3,
                                                     modsB + (size_t)c * BB_ * CC_,
                                                     escB + (size_t)c * BB_ * CC_);

    dim3 ggrid(MTOT / BM, NTOT / BN);  // 256 x 2
    // 4) input 1x1 conv: x = w0 @ e + b0 (no mod, no demod, no noise, no lrelu)
    gemm_conv_kernel<1, false, false, false, false><<<ggrid, 256, 0, stream>>>(
        e, wt0, nullptr, nullptr, b0, nullptr, xbuf, 0);
    // 5) residual translator blocks
    int eg = (int)((NEL + 255) / 256);
    for (int i = 0; i < LL_; ++i) {
        int c0 = i * 2 + 0, c1 = i * 2 + 1;
        gemm_conv_kernel<3, true, true, true, true><<<ggrid, 256, 0, stream>>>(
            xbuf, wtB + (size_t)c0 * CC_ * K3, modsB + (size_t)c0 * BB_ * CC_,
            escB + (size_t)c0 * BB_ * CC_, bb + (size_t)c0 * CC_, bnw + (size_t)c0 * CC_,
            hA, c0);
        gemm_conv_kernel<3, true, true, true, true><<<ggrid, 256, 0, stream>>>(
            hA, wtB + (size_t)c1 * CC_ * K3, modsB + (size_t)c1 * BB_ * CC_,
            escB + (size_t)c1 * BB_ * CC_, bb + (size_t)c1 * CC_, bnw + (size_t)c1 * CC_,
            hB, c1);
        residual_kernel<<<eg, 256, 0, stream>>>(xbuf, hB, (int)NEL);
    }
    // 6) heads: f -> d_out, mask-raw -> hA (free now)
    gemm_conv_kernel<1, true, false, false, false><<<ggrid, 256, 0, stream>>>(
        xbuf, wtF, modF, nullptr, bf, nullptr, out, 0);
    gemm_conv_kernel<1, true, false, false, false><<<ggrid, 256, 0, stream>>>(
        xbuf, wtM, modM, nullptr, bm, nullptr, hA, 0);
    // 7) blend: out = f*sigmoid(m) + e*(1-sigmoid(m))
    final_kernel<<<eg, 256, 0, stream>>>(e, hA, out, (int)NEL);
    (void)in_sizes; (void)n_in; (void)out_size; (void)ws_size;
}
